// MLPBlock_4801773437415
// MI455X (gfx1250) — matrix operand order repaired and checked by reading, not compiled, not run on hardware
//
#include <hip/hip_runtime.h>
#include <hip/hip_bf16.h>

typedef __attribute__((ext_vector_type(16))) _Float16 v16h;
typedef __attribute__((ext_vector_type(8)))  _Float16 v8h;
typedef __attribute__((ext_vector_type(4)))  _Float16 v4h;
typedef __attribute__((ext_vector_type(8)))  float    v8f;

union V16 { v16h v; v8h h[2]; };

__device__ __forceinline__ float gelu_tanh(float x) {
  // jax.nn.gelu(approximate=True) == x * sigmoid(2*sqrt(2/pi)*(x+0.044715x^3))
  const float c0 = 0.7978845608028654f;
  const float c1 = 0.044715f;
  float u = c0 * (x + c1 * x * x * x);
  float e = __expf(-2.0f * u);          // branch-free: v_exp_f32
  return x / (1.0f + e);                // u->-inf: e->inf -> 0 ; u->+inf: e->0 -> x
}

// ---- fp32 -> fp16 conversion (vectorized 4-wide, grid-stride) ----
__global__ void cvt_f32_to_f16_x4(const float4* __restrict__ in,
                                  v4h* __restrict__ out, long n4) {
  long i = (long)blockIdx.x * blockDim.x + threadIdx.x;
  long stride = (long)gridDim.x * blockDim.x;
  for (; i < n4; i += stride) {
    float4 f = in[i];
    v4h h;
    h.x = (_Float16)f.x; h.y = (_Float16)f.y;
    h.z = (_Float16)f.z; h.w = (_Float16)f.w;
    out[i] = h;
  }
}

// ---- Block-ELL layer: y[m, r*16+n] = sum_k x[m, cols[r,k]*16 + i] * W[r,k,n,i]
// Block = 4 waves, all on the same ELL row-block r. Weights for r are staged in
// LDS once (fc1 32KB, fc2 128KB; WGP LDS is 320KB) and served as B-fragments by
// ds_load_b128. Each wave owns MT=8 accumulator tiles (a 128-row M strip), so
// each B-frag feeds 8 back-to-back v_wmma_f32_16x16x32_f16 ops.
template <int KBLK, int MT, int IN_STRIDE, int OUT_STRIDE, bool GELU, typename OutT>
__global__ void __launch_bounds__(128)
block_ell_wmma(const _Float16* __restrict__ xin,   // [M, IN_STRIDE] f16
               const _Float16* __restrict__ wgt,   // [R, KBLK, 16, 16] f16 (o,i)
               const int*      __restrict__ cols,  // [R, KBLK]
               const float*    __restrict__ bias,  // [R*16]
               OutT*           __restrict__ yout)  // [M, OUT_STRIDE]
{
  extern __shared__ _Float16 smem[];               // KBLK * 256 halves

  const int lane    = threadIdx.x & 31;
  const int wave    = threadIdx.x >> 5;            // 4 waves / block
  const int r       = blockIdx.x;
  const int nIdx    = lane & 15;                   // row (A) / col (B,C) in tile
  const int halfSel = lane >> 4;                   // lane-group selects K-half
  const int k0      = halfSel * 8;
  const long m0     = ((long)blockIdx.y * 4 + wave) * (MT * 16);

  // Cooperative stage of this row-block's weights into LDS (b128 -> ds_store).
  {
    const v8h* __restrict__ wsrc = (const v8h*)(wgt + (long)r * KBLK * 256);
    v8h* wdst = (v8h*)smem;
    const int nvec = KBLK * 32;                    // 16B chunks
    for (int i = threadIdx.x; i < nvec; i += 128) wdst[i] = wsrc[i];
  }
  __syncthreads();

  const int* colr = cols + (long)r * KBLK;

  v8f acc[MT];
#pragma unroll
  for (int t = 0; t < MT; ++t) acc[t] = v8f{};

  for (int j = 0; j < KBLK; j += 2) {
    const int cA = colr[j];
    const int cB = colr[j + 1];

    // B fragment (32x16) from LDS: halves {k0..k0+7} of block j => K 0..15,
    // of block j+1 => K 16..31 (matches A fragment order).
    V16 b;
    const _Float16* wp = smem + (j * 16 + nIdx) * 16 + k0;
    b.h[0] = *(const v8h*)(wp);
    b.h[1] = *(const v8h*)(wp + 256);

    const long oA = (long)cA * 16 + k0;
    const long oB = (long)cB * 16 + k0;
    const _Float16* xr = xin + (m0 + nIdx) * (long)IN_STRIDE;

    // A fragment (16x32): row m=nIdx; halves {k0..k0+7} of block cA => K 0..15,
    // of block cB => K 16..31. Two 16B contiguous global loads per lane.
#pragma unroll
    for (int t = 0; t < MT; ++t) {
      V16 a;
      const _Float16* xt = xr + (long)t * 16 * IN_STRIDE;
      a.h[0] = *(const v8h*)(xt + oA);
      a.h[1] = *(const v8h*)(xt + oB);
      acc[t] = __builtin_amdgcn_wmma_f32_16x16x32_f16(
          /*neg_a=*/false, a.v, /*neg_b=*/false, b.v,
          /*c_mod=*/(short)0, acc[t], /*reuse_a=*/false, /*reuse_b=*/false);
    }
  }

  // Epilogue: C/D layout -> lane l owns N = l%16, rows M = d + 8*(l/16), d=0..7
  const float bv = bias[(long)r * 16 + nIdx];
  OutT* ybase = yout + (long)r * 16 + nIdx;
#pragma unroll
  for (int t = 0; t < MT; ++t) {
#pragma unroll
    for (int d = 0; d < 8; ++d) {
      long m = m0 + (long)t * 16 + d + 8 * halfSel;
      float v = acc[t][d] + bv;
      if (GELU) v = gelu_tanh(v);
      ybase[m * (long)OUT_STRIDE] = (OutT)v;
    }
  }
}

extern "C" void kernel_launch(void* const* d_in, const int* in_sizes, int n_in,
                              void* d_out, int out_size, void* d_ws, size_t ws_size,
                              hipStream_t stream) {
  const float* x  = (const float*)d_in[0];   // (4,2048,2048)
  const float* w1 = (const float*)d_in[1];   // (512,64,16,16)
  const int*   c1 = (const int*)d_in[2];     // (512,64)
  const float* b1 = (const float*)d_in[3];   // (8192,)
  const float* w2 = (const float*)d_in[4];   // (128,256,16,16)
  const int*   c2 = (const int*)d_in[5];     // (128,256)
  const float* b2 = (const float*)d_in[6];   // (2048,)
  float* out = (float*)d_out;

  const int d_model = 2048;
  const long n_x  = (long)in_sizes[0];
  const long n_w1 = (long)in_sizes[1];
  const long n_w2 = (long)in_sizes[4];
  const long M    = n_x / d_model;           // 8192 tokens

  // Workspace layout (256B-aligned): xh | w1h | w2h | hiddenh  (~192 MB total)
  char* ws = (char*)d_ws;
  size_t off = 0;
  _Float16* xh  = (_Float16*)(ws + off); off += (size_t)n_x  * 2; off = (off + 255) & ~(size_t)255;
  _Float16* w1h = (_Float16*)(ws + off); off += (size_t)n_w1 * 2; off = (off + 255) & ~(size_t)255;
  _Float16* w2h = (_Float16*)(ws + off); off += (size_t)n_w2 * 2; off = (off + 255) & ~(size_t)255;
  _Float16* hh  = (_Float16*)(ws + off); // M * d_ff halves

  // fp32 -> fp16 staging
  cvt_f32_to_f16_x4<<<4096, 256, 0, stream>>>((const float4*)x,  (v4h*)xh,  n_x  / 4);
  cvt_f32_to_f16_x4<<<2048, 256, 0, stream>>>((const float4*)w1, (v4h*)w1h, n_w1 / 4);
  cvt_f32_to_f16_x4<<<2048, 256, 0, stream>>>((const float4*)w2, (v4h*)w2h, n_w2 / 4);

  // fc1: (M,2048) x ELL(512 rblk, K=64) -> gelu -> hidden f16 (M,8192)
  dim3 g1(512u, (unsigned)(M / 512));
  block_ell_wmma<64, 8, 2048, 8192, true, _Float16>
      <<<g1, 128, 64 * 512, stream>>>(xh, w1h, c1, b1, hh);

  // fc2: (M,8192) x ELL(128 rblk, K=256) -> out f32 (M,2048)
  dim3 g2(128u, (unsigned)(M / 512));
  block_ell_wmma<256, 8, 8192, 2048, false, float>
      <<<g2, 128, 256 * 512, stream>>>(hh, w2h, c2, b2, out);
}
